// ConformerBlock_68118181314709
// MI455X (gfx1250) — 16-bit and 8-bit matrix operand order repaired and checked by reading, compiled, not run on hardware
//
#include <hip/hip_runtime.h>
#include <stdint.h>

typedef __attribute__((ext_vector_type(8)))  int      v8i;
typedef __attribute__((ext_vector_type(8)))  float    v8f;
typedef __attribute__((ext_vector_type(16))) _Float16 v16h;

#define NB  4
#define TT  2048
#define DD  512
#define NH  8
#define HD  64
#define NN  (NB*TT)
#define D3  (3*DD)
#define D4  (4*DD)
#define KCV 31

__device__ __forceinline__ float redmax16(float v) {
#pragma unroll
  for (int m = 1; m < 16; m <<= 1) v = fmaxf(v, __shfl_xor(v, m, 32));
  return v;
}
__device__ __forceinline__ float redsum16(float v) {
#pragma unroll
  for (int m = 1; m < 16; m <<= 1) v += __shfl_xor(v, m, 32);
  return v;
}

__global__ void k_absmean_partial(const float* __restrict__ w, int n, float* __restrict__ part) {
  __shared__ float red[256];
  float s = 0.f;
  for (int i = blockIdx.x * blockDim.x + threadIdx.x; i < n; i += gridDim.x * blockDim.x)
    s += fabsf(w[i]);
  red[threadIdx.x] = s; __syncthreads();
  for (int st = 128; st > 0; st >>= 1) {
    if (threadIdx.x < st) red[threadIdx.x] += red[threadIdx.x + st];
    __syncthreads();
  }
  if (threadIdx.x == 0) part[blockIdx.x] = red[0];
}

__global__ void k_absmean_final(const float* __restrict__ part, int n, float* __restrict__ scal) {
  __shared__ float red[256];
  red[threadIdx.x] = part[threadIdx.x];
  __syncthreads();
  for (int st = 128; st > 0; st >>= 1) {
    if (threadIdx.x < st) red[threadIdx.x] += red[threadIdx.x + st];
    __syncthreads();
  }
  if (threadIdx.x == 0) scal[0] = fmaxf(red[0] / (float)n, 1e-5f);
}

__global__ void k_wquant(const float* __restrict__ w, int8_t* __restrict__ wq, int n,
                         const float* __restrict__ scal) {
  int i = blockIdx.x * blockDim.x + threadIdx.x;
  if (i >= n) return;
  float inv = 1.f / scal[0];
  wq[i] = (int8_t)fminf(fmaxf(rintf(w[i] * inv), -1.f), 1.f);
}

__global__ void k_rmsnorm_quant(const float* __restrict__ x, const float* __restrict__ nw,
                                int8_t* __restrict__ q8, float* __restrict__ rs) {
  const int row = blockIdx.x, tid = threadIdx.x;
  const float* xr = x + (size_t)row * DD;
  __shared__ float red[256];
  float v0 = xr[tid], v1 = xr[tid + 256];
  red[tid] = v0 * v0 + v1 * v1; __syncthreads();
  for (int st = 128; st > 0; st >>= 1) { if (tid < st) red[tid] += red[tid + st]; __syncthreads(); }
  float rms = rsqrtf(red[0] / (float)DD + 1e-6f);
  __syncthreads();
  float a0 = v0 * rms * nw[tid], a1 = v1 * rms * nw[tid + 256];
  red[tid] = fmaxf(fabsf(a0), fabsf(a1)); __syncthreads();
  for (int st = 128; st > 0; st >>= 1) { if (tid < st) red[tid] = fmaxf(red[tid], red[tid + st]); __syncthreads(); }
  float amax = fmaxf(red[0], 1e-5f);
  float s = 127.f / amax;
  int8_t* qr = q8 + (size_t)row * DD;
  qr[tid]       = (int8_t)fminf(fmaxf(rintf(a0 * s), -128.f), 127.f);
  qr[tid + 256] = (int8_t)fminf(fmaxf(rintf(a1 * s), -128.f), 127.f);
  if (tid == 0) rs[row] = amax / 127.f;
}

__global__ void k_rmsnorm_f16(const float* __restrict__ x, const float* __restrict__ nw,
                              _Float16* __restrict__ out) {
  const int row = blockIdx.x, tid = threadIdx.x;
  const float* xr = x + (size_t)row * DD;
  __shared__ float red[256];
  float v0 = xr[tid], v1 = xr[tid + 256];
  red[tid] = v0 * v0 + v1 * v1; __syncthreads();
  for (int st = 128; st > 0; st >>= 1) { if (tid < st) red[tid] += red[tid + st]; __syncthreads(); }
  float rms = rsqrtf(red[0] / (float)DD + 1e-6f);
  _Float16* orow = out + (size_t)row * DD;
  orow[tid]       = (_Float16)(v0 * rms * nw[tid]);
  orow[tid + 256] = (_Float16)(v1 * rms * nw[tid + 256]);
}

__global__ void k_rmsnorm_out(const float* __restrict__ x, const float* __restrict__ nw,
                              float* __restrict__ out) {
  const int row = blockIdx.x, tid = threadIdx.x;
  const float* xr = x + (size_t)row * DD;
  __shared__ float red[256];
  float v0 = xr[tid], v1 = xr[tid + 256];
  red[tid] = v0 * v0 + v1 * v1; __syncthreads();
  for (int st = 128; st > 0; st >>= 1) { if (tid < st) red[tid] += red[tid + st]; __syncthreads(); }
  float rms = rsqrtf(red[0] / (float)DD + 1e-6f);
  float* orow = out + (size_t)row * DD;
  orow[tid]       = v0 * rms * nw[tid];
  orow[tid + 256] = v1 * rms * nw[tid + 256];
}

__global__ void k_snake_quant(const float* __restrict__ h, const float* __restrict__ la,
                              const float* __restrict__ lb, int8_t* __restrict__ q8,
                              float* __restrict__ rs, int Dh) {
  const int row = blockIdx.x, tid = threadIdx.x;
  const float* hr = h + (size_t)row * Dh;
  const int cnt = Dh >> 8;
  float tmp[8];
  __shared__ float red[256];
  float am = 0.f;
  for (int e = 0; e < cnt; ++e) {
    int c = e * 256 + tid;
    float v = hr[c];
    float a = __expf(la[c]), b = __expf(lb[c]);
    float sn = __sinf(a * v);
    float o = v + sn * sn / (b + 1e-9f);
    tmp[e] = o;
    am = fmaxf(am, fabsf(o));
  }
  red[tid] = am; __syncthreads();
  for (int st = 128; st > 0; st >>= 1) { if (tid < st) red[tid] = fmaxf(red[tid], red[tid + st]); __syncthreads(); }
  float amax = fmaxf(red[0], 1e-5f);
  float s = 127.f / amax;
  int8_t* qr = q8 + (size_t)row * Dh;
  for (int e = 0; e < cnt; ++e) {
    int c = e * 256 + tid;
    qr[c] = (int8_t)fminf(fmaxf(rintf(tmp[e] * s), -128.f), 127.f);
  }
  if (tid == 0) rs[row] = amax / 127.f;
}

__global__ void k_gemm_i8(const int8_t* __restrict__ A, const int8_t* __restrict__ W,
                          const float* __restrict__ rs, const float* __restrict__ wscal,
                          float* __restrict__ out, int M, int Kd, int Nd) {
  const int lane = threadIdx.x & 31, wave = threadIdx.x >> 5;
  const int half = lane >> 4, l16 = lane & 15;
  const int tilesN4 = Nd >> 6;
  int tile = blockIdx.x * 8 + wave;
  int tm = tile / tilesN4, tg = tile % tilesN4;
  if (tm >= (M >> 4)) return;
  const int8_t* Arow  = A + (size_t)(tm * 16 + l16) * Kd;
  const int8_t* Wrow0 = W + (size_t)(tg * 64 + l16) * Kd;
  v8i acc[4] = {};
  for (int k0 = 0; k0 < Kd; k0 += 64) {
    __builtin_prefetch(Arow + k0 + 64, 0, 0);
    v8i a;
#pragma unroll
    for (int v = 0; v < 8; ++v) {
      int ks = (v >> 1) * 16 + half * 8 + (v & 1) * 4;
      a[v] = *(const int*)(Arow + k0 + ks);
    }
#pragma unroll
    for (int j = 0; j < 4; ++j) {
      const int8_t* Wrow = Wrow0 + (size_t)(j * 16) * Kd;
      __builtin_prefetch(Wrow + k0 + 64, 0, 0);
      v8i b;
#pragma unroll
      for (int v = 0; v < 8; ++v) {
        int ks = (v >> 1) * 16 + half * 8 + (v & 1) * 4;
        b[v] = *(const int*)(Wrow + k0 + ks);
      }
      acc[j] = __builtin_amdgcn_wmma_i32_16x16x64_iu8(true, a, true, b, acc[j], false, false);
    }
  }
  const float sw = wscal[0];
#pragma unroll
  for (int r = 0; r < 8; ++r) {
    int m = tm * 16 + r + half * 8;
    float rv = rs[m] * sw;
#pragma unroll
    for (int j = 0; j < 4; ++j)
      out[(size_t)m * Nd + tg * 64 + j * 16 + l16] = (float)acc[j][r] * rv;
  }
}

__global__ void k_gemm_f16(const _Float16* __restrict__ A, const _Float16* __restrict__ W,
                           const float* __restrict__ bias, _Float16* __restrict__ outh,
                           float* __restrict__ outf, int M, int Kd, int Nd, int addres) {
  const int lane = threadIdx.x & 31, wave = threadIdx.x >> 5;
  const int half = lane >> 4, l16 = lane & 15;
  const int tilesN4 = Nd >> 6;
  int tile = blockIdx.x * 8 + wave;
  int tm = tile / tilesN4, tg = tile % tilesN4;
  if (tm >= (M >> 4)) return;
  const _Float16* Arow  = A + (size_t)(tm * 16 + l16) * Kd;
  const _Float16* Wrow0 = W + (size_t)(tg * 64 + l16) * Kd;
  v8f acc[4] = {};
  for (int k0 = 0; k0 < Kd; k0 += 32) {
    __builtin_prefetch(Arow + k0 + 32, 0, 0);
    v16h a;
#pragma unroll
    for (int v = 0; v < 8; ++v) {
      int kk = k0 + (v >> 2) * 16 + half * 8 + (v & 3) * 2;
      a[2 * v] = Arow[kk]; a[2 * v + 1] = Arow[kk + 1];
    }
#pragma unroll
    for (int j = 0; j < 4; ++j) {
      const _Float16* Wrow = Wrow0 + (size_t)(j * 16) * Kd;
      __builtin_prefetch(Wrow + k0 + 32, 0, 0);
      v16h b;
#pragma unroll
      for (int v = 0; v < 8; ++v) {
        int kk = k0 + (v >> 2) * 16 + half * 8 + (v & 3) * 2;
        b[2 * v] = Wrow[kk]; b[2 * v + 1] = Wrow[kk + 1];
      }
      acc[j] = __builtin_amdgcn_wmma_f32_16x16x32_f16(false, a, false, b, (short)0, acc[j], false, false);
    }
  }
#pragma unroll
  for (int r = 0; r < 8; ++r) {
    int m = tm * 16 + r + half * 8;
#pragma unroll
    for (int j = 0; j < 4; ++j) {
      int n = tg * 64 + j * 16 + l16;
      float val = acc[j][r] + bias[n];
      if (addres) outf[(size_t)m * Nd + n] += val;
      else        outh[(size_t)m * Nd + n] = (_Float16)val;
    }
  }
}

__global__ void k_flash(const _Float16* __restrict__ qkv, _Float16* __restrict__ o) {
  const int lane = threadIdx.x & 31, wave = threadIdx.x >> 5;
  const int half = lane >> 4, l16 = lane & 15;
  __shared__ _Float16 Pl[8][16][32];
  int gw = blockIdx.x * 8 + wave;
  const int QT = TT / 16;
  int qt = gw % QT;
  int bh = gw / QT;
  int h = bh % NH, b = bh / NH;
  const _Float16* Qb = qkv + (size_t)b * TT * D3 + h * HD;
  const _Float16* Kb = Qb + DD;
  const _Float16* Vb = Qb + 2 * DD;

  v16h qf[2];
  {
    const _Float16* qr = Qb + (size_t)(qt * 16 + l16) * D3;
#pragma unroll
    for (int kc = 0; kc < 2; ++kc)
#pragma unroll
      for (int v = 0; v < 8; ++v) {
        int kk = kc * 32 + (v >> 2) * 16 + half * 8 + (v & 3) * 2;
        qf[kc][2 * v] = qr[kk]; qf[kc][2 * v + 1] = qr[kk + 1];
      }
  }
  float mrow[8], lrow[8];
#pragma unroll
  for (int r = 0; r < 8; ++r) { mrow[r] = -3.0e38f; lrow[r] = 0.f; }
  v8f oacc[4] = {};
  const float scale = 0.125f;

  for (int kt = 0; kt < TT / 32; ++kt) {
    v8f s0 = {}, s1 = {};
#pragma unroll
    for (int st = 0; st < 2; ++st) {
      v8f acc = {};
      const _Float16* kr = Kb + (size_t)(kt * 32 + st * 16 + l16) * D3;
#pragma unroll
      for (int kc = 0; kc < 2; ++kc) {
        v16h bf;
#pragma unroll
        for (int v = 0; v < 8; ++v) {
          int kk = kc * 32 + (v >> 2) * 16 + half * 8 + (v & 3) * 2;
          bf[2 * v] = kr[kk]; bf[2 * v + 1] = kr[kk + 1];
        }
        acc = __builtin_amdgcn_wmma_f32_16x16x32_f16(false, qf[kc], false, bf, (short)0, acc, false, false);
      }
      if (st == 0) s0 = acc; else s1 = acc;
    }
#pragma unroll
    for (int r = 0; r < 8; ++r) {
      float v0 = s0[r] * scale, v1 = s1[r] * scale;
      float nm = fmaxf(mrow[r], redmax16(fmaxf(v0, v1)));
      float f = __expf(mrow[r] - nm);
      float p0 = __expf(v0 - nm), p1 = __expf(v1 - nm);
      lrow[r] = lrow[r] * f + redsum16(p0 + p1);
      mrow[r] = nm;
#pragma unroll
      for (int j = 0; j < 4; ++j) oacc[j][r] *= f;
      Pl[wave][r + half * 8][l16]      = (_Float16)p0;
      Pl[wave][r + half * 8][16 + l16] = (_Float16)p1;
    }
    __builtin_amdgcn_wave_barrier();
    v16h pa;
    const _Float16* pr = &Pl[wave][l16][0];
#pragma unroll
    for (int v = 0; v < 8; ++v) {
      int kk = (v >> 2) * 16 + half * 8 + (v & 3) * 2;
      pa[2 * v] = pr[kk]; pa[2 * v + 1] = pr[kk + 1];
    }
#pragma unroll
    for (int j = 0; j < 4; ++j) {
      v16h vf;
#pragma unroll
      for (int v = 0; v < 8; ++v) {
        const _Float16* vr = Vb + (size_t)(kt * 32 + (v >> 2) * 16 + half * 8 + (v & 3) * 2) * D3 + j * 16 + l16;
        vf[2 * v] = vr[0]; vf[2 * v + 1] = vr[D3];
      }
      oacc[j] = __builtin_amdgcn_wmma_f32_16x16x32_f16(false, pa, false, vf, (short)0, oacc[j], false, false);
    }
  }
#pragma unroll
  for (int r = 0; r < 8; ++r) {
    float inv = 1.f / lrow[r];
    int row = b * TT + qt * 16 + r + half * 8;
#pragma unroll
    for (int j = 0; j < 4; ++j)
      o[(size_t)row * DD + h * HD + j * 16 + l16] = (_Float16)(oacc[j][r] * inv);
  }
}

__global__ void k_glu(const float* __restrict__ h, float* __restrict__ out) {
  int i = blockIdx.x * blockDim.x + threadIdx.x;
  int m = i / DD, c = i % DD;
  float a = h[(size_t)m * (2 * DD) + c];
  float g = h[(size_t)m * (2 * DD) + DD + c];
  out[i] = a * (1.f / (1.f + __expf(-g)));
}

__global__ void k_dwconv_bn(const float* __restrict__ in, const float* __restrict__ w,
                            const float* __restrict__ wb, const float* __restrict__ g,
                            const float* __restrict__ bb, const float* __restrict__ bm,
                            const float* __restrict__ bv, float* __restrict__ out) {
  int i = blockIdx.x * blockDim.x + threadIdx.x;
  int d = i % DD;
  int t = (i / DD) % TT;
  int b = i / (DD * TT);
  const float* base = in + (size_t)b * TT * DD + d;
  float s = 0.f;
#pragma unroll
  for (int k = 0; k < KCV; ++k) {
    int tt = t + k - (KCV / 2);
    if (tt >= 0 && tt < TT) s += base[(size_t)tt * DD] * w[d * KCV + k];
  }
  s += wb[d];
  s = (s - bm[d]) * rsqrtf(bv[d] + 1e-5f) * g[d] + bb[d];
  out[i] = s;
}

__global__ void k_addres(float* __restrict__ x, const float* __restrict__ y, float alpha) {
  int i = blockIdx.x * blockDim.x + threadIdx.x;
  x[i] += alpha * y[i];
}

__global__ void k_addres_init(const float* __restrict__ a, const float* __restrict__ y,
                              float alpha, float* __restrict__ xo) {
  int i = blockIdx.x * blockDim.x + threadIdx.x;
  xo[i] = a[i] + alpha * y[i];
}

__global__ void k_f32tof16(const float* __restrict__ w, _Float16* __restrict__ o, int n) {
  int i = blockIdx.x * blockDim.x + threadIdx.x;
  if (i < n) o[i] = (_Float16)w[i];
}

extern "C" void kernel_launch(void* const* d_in, const int* in_sizes, int n_in,
                              void* d_out, int out_size, void* d_ws, size_t ws_size,
                              hipStream_t stream) {
  (void)in_sizes; (void)n_in; (void)out_size; (void)ws_size;
  const float* x       = (const float*)d_in[0];
  const float* ff1_nw  = (const float*)d_in[1];
  const float* ff1_w1  = (const float*)d_in[2];
  const float* ff1_a   = (const float*)d_in[3];
  const float* ff1_b   = (const float*)d_in[4];
  const float* ff1_w2  = (const float*)d_in[5];
  const float* attn_nw = (const float*)d_in[6];
  const float* ipw     = (const float*)d_in[7];
  const float* ipb     = (const float*)d_in[8];
  const float* opw     = (const float*)d_in[9];
  const float* opb     = (const float*)d_in[10];
  const float* conv_nw = (const float*)d_in[11];
  const float* pw1     = (const float*)d_in[12];
  const float* dww     = (const float*)d_in[13];
  const float* dwb     = (const float*)d_in[14];
  const float* bng     = (const float*)d_in[15];
  const float* bnb     = (const float*)d_in[16];
  const float* bnm     = (const float*)d_in[17];
  const float* bnv     = (const float*)d_in[18];
  const float* ska     = (const float*)d_in[19];
  const float* skb     = (const float*)d_in[20];
  const float* pw2     = (const float*)d_in[21];
  const float* ff2_nw  = (const float*)d_in[22];
  const float* ff2_w1  = (const float*)d_in[23];
  const float* ff2_a   = (const float*)d_in[24];
  const float* ff2_b   = (const float*)d_in[25];
  const float* ff2_w2  = (const float*)d_in[26];
  const float* fin_nw  = (const float*)d_in[27];

  char* ws = (char*)d_ws;
  size_t off = 0;
  auto alloc = [&](size_t bytes) {
    char* p = ws + off;
    off += (bytes + 255) & ~(size_t)255;
    return p;
  };
  float*    X    = (float*)alloc((size_t)NN * DD * 4);
  float*    Y    = (float*)alloc((size_t)NN * DD * 4);
  float*    Hb   = (float*)alloc((size_t)NN * D4 * 4);
  int8_t*   Q8   = (int8_t*)alloc((size_t)NN * D4);
  float*    RS   = (float*)alloc((size_t)NN * 4);
  _Float16* QKV  = (_Float16*)alloc((size_t)NN * D3 * 2);
  _Float16* F16A = (_Float16*)alloc((size_t)NN * DD * 2);
  int8_t*   W8   = (int8_t*)alloc((size_t)D4 * DD);
  _Float16* WF16 = (_Float16*)alloc((size_t)D3 * DD * 2);
  float*    WSC  = (float*)alloc(256);
  float*    PART = (float*)alloc(256 * 4);

  auto quantw = [&](const float* w, int n) {
    k_absmean_partial<<<256, 256, 0, stream>>>(w, n, PART);
    k_absmean_final<<<1, 256, 0, stream>>>(PART, n, WSC);
    k_wquant<<<(n + 255) / 256, 256, 0, stream>>>(w, W8, n, WSC);
  };
  auto gemm_i8 = [&](float* out, int Kd, int Nd) {
    int tiles = (NN / 16) * (Nd / 64);
    k_gemm_i8<<<tiles / 8, 256, 0, stream>>>(Q8, W8, RS, WSC, out, NN, Kd, Nd);
  };
  const int elems = NN * DD;

  k_rmsnorm_quant<<<NN, 256, 0, stream>>>(x, ff1_nw, Q8, RS);
  quantw(ff1_w1, D4 * DD);
  gemm_i8(Hb, DD, D4);
  k_snake_quant<<<NN, 256, 0, stream>>>(Hb, ff1_a, ff1_b, Q8, RS, D4);
  quantw(ff1_w2, DD * D4);
  gemm_i8(Y, D4, DD);
  k_addres_init<<<elems / 256, 256, 0, stream>>>(x, Y, 0.5f, X);

  k_rmsnorm_f16<<<NN, 256, 0, stream>>>(X, attn_nw, F16A);
  k_f32tof16<<<(D3 * DD) / 256, 256, 0, stream>>>(ipw, WF16, D3 * DD);
  k_gemm_f16<<<((NN / 16) * (D3 / 64)) / 8, 256, 0, stream>>>(
      F16A, WF16, ipb, QKV, nullptr, NN, DD, D3, 0);
  k_flash<<<(NB * NH * (TT / 16)) / 8, 256, 0, stream>>>(QKV, F16A);
  k_f32tof16<<<(DD * DD) / 256, 256, 0, stream>>>(opw, WF16, DD * DD);
  k_gemm_f16<<<((NN / 16) * (DD / 64)) / 8, 256, 0, stream>>>(
      F16A, WF16, opb, nullptr, X, NN, DD, DD, 1);

  k_rmsnorm_quant<<<NN, 256, 0, stream>>>(X, conv_nw, Q8, RS);
  quantw(pw1, 2 * DD * DD);
  gemm_i8(Hb, DD, 2 * DD);
  k_glu<<<elems / 256, 256, 0, stream>>>(Hb, Y);
  k_dwconv_bn<<<elems / 256, 256, 0, stream>>>(Y, dww, dwb, bng, bnb, bnm, bnv, Hb);
  k_snake_quant<<<NN, 256, 0, stream>>>(Hb, ska, skb, Q8, RS, DD);
  quantw(pw2, DD * DD);
  gemm_i8(Y, DD, DD);
  k_addres<<<elems / 256, 256, 0, stream>>>(X, Y, 1.0f);

  k_rmsnorm_quant<<<NN, 256, 0, stream>>>(X, ff2_nw, Q8, RS);
  quantw(ff2_w1, D4 * DD);
  gemm_i8(Hb, DD, D4);
  k_snake_quant<<<NN, 256, 0, stream>>>(Hb, ff2_a, ff2_b, Q8, RS, D4);
  quantw(ff2_w2, DD * D4);
  gemm_i8(Y, D4, DD);
  k_addres<<<elems / 256, 256, 0, stream>>>(X, Y, 0.5f);

  k_rmsnorm_out<<<NN, 256, 0, stream>>>(X, fin_nw, (float*)d_out);
}
